// ChildSumTreeLSTMEncoder_54365696033410
// MI455X (gfx1250) — matrix operand order repaired and checked by reading, not compiled, not run on hardware
//
#include <hip/hip_runtime.h>
#include <hip/hip_bf16.h>

// ---------------------------------------------------------------------------
// Child-Sum Tree-LSTM, bottom-up over 6 levels. GEMMs run on CDNA5 WMMA
// (v_wmma_f32_16x16x32_bf16, f32 accumulate). V=50000 E=300 H=512 L=6 N=8192 K=4
// GEMM: double-buffered LDS + pipelined B-fragments.
// ---------------------------------------------------------------------------

typedef __attribute__((ext_vector_type(16))) __bf16 v16bf;
typedef __attribute__((ext_vector_type(8)))  float  v8f;
typedef __attribute__((ext_vector_type(4)))  int    i32x4;

#define NN   8192
#define EE   300
#define EPAD 320
#define HH   512
#define KK   4

// ------------------------- weight transpose + bf16 convert ------------------
// W [K, Ncols] f32 (row-major)  ->  WT [Ncols, Kpad] bf16 (row-major), zero pad
__global__ void transpose_convert_kernel(const float* __restrict__ W,
                                         __bf16* __restrict__ WT,
                                         int K, int Ncols, int Kpad) {
    int idx = blockIdx.x * blockDim.x + threadIdx.x;
    int n = idx / Kpad;
    int k = idx - n * Kpad;
    if (n >= Ncols) return;
    float v = (k < K) ? W[(size_t)k * Ncols + n] : 0.0f;
    WT[(size_t)n * Kpad + k] = (__bf16)v;
}

// ------------------------- embedding gather -> bf16 --------------------------
__global__ void gather_embed_kernel(const int* __restrict__ vocab_ix,
                                    const float* __restrict__ token_mask,
                                    const float* __restrict__ embed,
                                    __bf16* __restrict__ X) {
    int n = blockIdx.x;
    int ix = vocab_ix[n];
    float tm = token_mask[n];
    const float* row = embed + (size_t)ix * EE;
    for (int e = threadIdx.x; e < EPAD; e += blockDim.x) {
        float v = (e < EE) ? row[e] * tm : 0.0f;
        X[(size_t)n * EPAD + e] = (__bf16)v;
    }
}

// ------------------------- child gather: h_ch (bf16) and h_tilde (bf16) ------
__global__ void gather_children_kernel(const int* __restrict__ child_idx,
                                       const float* __restrict__ child_mask,
                                       const float* __restrict__ h_prev,
                                       __bf16* __restrict__ hch,
                                       __bf16* __restrict__ htilde) {
    int n = blockIdx.x;
    __shared__ int   sidx[KK];
    __shared__ float smsk[KK];
    if (threadIdx.x < KK) {
        sidx[threadIdx.x] = child_idx[n * KK + threadIdx.x];
        smsk[threadIdx.x] = child_mask[n * KK + threadIdx.x];
    }
    __syncthreads();
    for (int h = threadIdx.x; h < HH; h += blockDim.x) {
        float s = 0.0f;
#pragma unroll
        for (int k = 0; k < KK; ++k) {
            float v = h_prev[(size_t)sidx[k] * HH + h] * smsk[k];
            hch[((size_t)n * KK + k) * HH + h] = (__bf16)v;
            s += v;
        }
        htilde[(size_t)n * HH + h] = (__bf16)s;
    }
}

// ------- fused forget gates: fc = sum_k sigmoid(xWf + hU_f[k] + b_f) * c_ch ---
__global__ void fuse_fc_kernel(const int* __restrict__ child_idx,
                               const float* __restrict__ child_mask,
                               const float* __restrict__ c_prev,
                               const float* __restrict__ xWf,
                               const float* __restrict__ fpre,
                               const float* __restrict__ b_f,
                               float* __restrict__ fc) {
    int n = blockIdx.x;
    __shared__ int   sidx[KK];
    __shared__ float smsk[KK];
    if (threadIdx.x < KK) {
        sidx[threadIdx.x] = child_idx[n * KK + threadIdx.x];
        smsk[threadIdx.x] = child_mask[n * KK + threadIdx.x];
    }
    __syncthreads();
    for (int h = threadIdx.x; h < HH; h += blockDim.x) {
        float xw = xWf[(size_t)n * HH + h];
        float bf = b_f[h];
        float acc = 0.0f;
#pragma unroll
        for (int k = 0; k < KK; ++k) {
            float cch = c_prev[(size_t)sidx[k] * HH + h] * smsk[k];
            float pre = xw + fpre[((size_t)n * KK + k) * HH + h] + bf;
            float f = 1.0f / (1.0f + __expf(-pre));
            acc += f * cch;
        }
        fc[(size_t)n * HH + h] = acc;
    }
}

// ------------------------- gate nonlinearity ---------------------------------
__global__ void gates_kernel(const float* __restrict__ iou,
                             const float* __restrict__ fc,
                             float* __restrict__ c_out,
                             float* __restrict__ h_out) {
    int idx = blockIdx.x * blockDim.x + threadIdx.x;   // over N*H
    int n = idx >> 9;
    int h = idx & (HH - 1);
    const float* row = iou + (size_t)n * (3 * HH);
    float i = row[h], o = row[HH + h], u = row[2 * HH + h];
    float si = 1.0f / (1.0f + __expf(-i));
    float so = 1.0f / (1.0f + __expf(-o));
    float c = si * tanhf(u) + fc[idx];
    c_out[idx] = c;
    h_out[idx] = so * tanhf(c);
}

__global__ void zero_f32_kernel(float* __restrict__ p, int nElems) {
    for (int i = blockIdx.x * blockDim.x + threadIdx.x; i < nElems;
         i += gridDim.x * blockDim.x)
        p[i] = 0.0f;
}

// ---------------------------------------------------------------------------
// WMMA GEMM:  C[M,N] = A[M,K] @ B[K,N]  (+ bias[N]) (+= existing C if addC)
// A row-major bf16 (stride K), BT = B^T row-major bf16 [N,K] (stride K),
// C row-major f32. M,N multiples of 128; K multiple of 32 (>= 64).
// Block: 256 threads = 8 waves; block tile 128x128; wave tile 16x128.
// Double-buffered LDS; global loads for chunk i+1 overlap WMMAs of chunk i;
// B fragments are register double-buffered so ds latency hides under WMMA.
// ---------------------------------------------------------------------------
#define LDS_STRIDE 40   // 32 K-halves + 8 pad halves -> 80B rows (16B aligned)

union Frag { v16bf v; i32x4 q[2]; };

__global__ void __launch_bounds__(256)
gemm_bf16_wmma_kernel(const __bf16* __restrict__ A,
                      const __bf16* __restrict__ BT,
                      float* __restrict__ C,
                      const float* __restrict__ bias,
                      int M, int Ncols, int K, int addC) {
    __shared__ __bf16 As[2][128 * LDS_STRIDE];
    __shared__ __bf16 Bs[2][128 * LDS_STRIDE];

    const int tid  = threadIdx.x;
    const int wave = tid >> 5;
    const int lane = tid & 31;
    const int l16  = lane & 15;
    const int hl   = lane >> 4;       // lane half: selects K sub-ranges
    const int bM   = blockIdx.x * 128;
    const int bN   = blockIdx.y * 128;

    v8f zero = {0.f, 0.f, 0.f, 0.f, 0.f, 0.f, 0.f, 0.f};
    v8f acc[8];
#pragma unroll
    for (int t = 0; t < 8; ++t) acc[t] = zero;

    // cooperative tile loads: thread t -> row t/2, half-row (t&1)*16
    const int r  = tid >> 1;
    const int ks = (tid & 1) << 4;
    const __bf16* gA = A  + (size_t)(bM + r) * K + ks;
    const __bf16* gB = BT + (size_t)(bN + r) * K + ks;
    const int sOff = r * LDS_STRIDE + ks;

    // per-wave fragment base offsets (within a buffer)
    const int aOff = (wave * 16 + l16) * LDS_STRIDE + 16 * hl;  // + {0,+8} later
    const int bOff = l16 * LDS_STRIDE + 16 * hl;                // + t*16*LDS_STRIDE

    // ---- prologue: stage chunk 0 into LDS buffer 0 ----
    i32x4 ra0 = *(const i32x4*)(gA);
    i32x4 ra1 = *(const i32x4*)(gA + 8);
    i32x4 rb0 = *(const i32x4*)(gB);
    i32x4 rb1 = *(const i32x4*)(gB + 8);
    *(i32x4*)&As[0][sOff]     = ra0;
    *(i32x4*)&As[0][sOff + 8] = ra1;
    *(i32x4*)&Bs[0][sOff]     = rb0;
    *(i32x4*)&Bs[0][sOff + 8] = rb1;

    const int nk = K >> 5;
    for (int i = 0; i < nk; ++i) {
        __syncthreads();                     // buffer (i&1) ready for all waves
        const int cur = i & 1;
        const int nxt = cur ^ 1;

        // issue global loads for next chunk early (overlap with WMMAs)
        if (i + 1 < nk) {
            const __bf16* pA = gA + (i + 1) * 32;
            const __bf16* pB = gB + (i + 1) * 32;
            ra0 = *(const i32x4*)(pA);
            ra1 = *(const i32x4*)(pA + 8);
            rb0 = *(const i32x4*)(pB);
            rb1 = *(const i32x4*)(pB + 8);
        }

        // A fragment (16x32): half 0 -> K 0-15, half 1 -> K 16-31
        Frag ua;
        {
            const __bf16* ab = &As[cur][aOff];
            ua.q[0] = *(const i32x4*)(ab);
            ua.q[1] = *(const i32x4*)(ab + 8);
        }
        // B fragments, register double-buffered across the 8 column tiles
        Frag ub0, ub1;
        {
            const __bf16* bb = &Bs[cur][bOff];
            ub0.q[0] = *(const i32x4*)(bb);
            ub0.q[1] = *(const i32x4*)(bb + 8);
        }
#pragma unroll
        for (int t = 0; t < 8; ++t) {
            Frag& cb = (t & 1) ? ub1 : ub0;
            Frag& nb = (t & 1) ? ub0 : ub1;
            if (t < 7) {
                const __bf16* bb = &Bs[cur][bOff + (t + 1) * 16 * LDS_STRIDE];
                nb.q[0] = *(const i32x4*)(bb);
                nb.q[1] = *(const i32x4*)(bb + 8);
            }
            acc[t] = __builtin_amdgcn_wmma_f32_16x16x32_bf16(
                false, ua.v, false, cb.v, (short)0, acc[t], false, false);
        }

        // drain staged global data into the other LDS buffer
        if (i + 1 < nk) {
            *(i32x4*)&As[nxt][sOff]     = ra0;
            *(i32x4*)&As[nxt][sOff + 8] = ra1;
            *(i32x4*)&Bs[nxt][sOff]     = rb0;
            *(i32x4*)&Bs[nxt][sOff + 8] = rb1;
        }
    }

    // Epilogue. C/D layout: VGPR j -> M = j (+8 for lane half 1), N = lane%16
    const int rowBase = bM + wave * 16 + hl * 8;
#pragma unroll
    for (int t = 0; t < 8; ++t) {
        int col = bN + t * 16 + l16;
        float b = (bias != nullptr) ? bias[col] : 0.0f;
#pragma unroll
        for (int j = 0; j < 8; ++j) {
            size_t off = (size_t)(rowBase + j) * Ncols + col;
            float v = acc[t][j] + b;
            if (addC) v += C[off];
            C[off] = v;
        }
    }
}

// ---------------------------------------------------------------------------
extern "C" void kernel_launch(void* const* d_in, const int* in_sizes, int n_in,
                              void* d_out, int out_size, void* d_ws, size_t ws_size,
                              hipStream_t stream) {
    (void)in_sizes; (void)n_in; (void)out_size; (void)ws_size;

    const int*   vocab_ix   = (const int*)d_in[0];    // [6, 8192]
    const int*   child_idx  = (const int*)d_in[1];    // [6, 8192, 4]
    const float* token_mask = (const float*)d_in[2];  // [6, 8192]
    const float* child_mask = (const float*)d_in[3];  // [6, 8192, 4]
    const float* embed      = (const float*)d_in[4];  // [V, 300]
    const float* W_iou      = (const float*)d_in[5];  // [300, 1536]
    const float* U_iou      = (const float*)d_in[6];  // [512, 1536]
    const float* b_iou      = (const float*)d_in[7];  // [1536]
    const float* W_f        = (const float*)d_in[8];  // [300, 512]
    const float* U_f        = (const float*)d_in[9];  // [512, 512]
    const float* b_f        = (const float*)d_in[10]; // [512]

    // ---- bump allocator over d_ws ----
    size_t off = 0;
    char* base = (char*)d_ws;
    auto take = [&](size_t bytes) -> void* {
        void* p = base + off;
        off += (bytes + 255) & ~(size_t)255;
        return p;
    };
    __bf16* Xbf   = (__bf16*)take((size_t)NN * EPAD * 2);          // [8192,320]
    __bf16* WiouT = (__bf16*)take((size_t)1536 * EPAD * 2);        // [1536,320]
    __bf16* WfT   = (__bf16*)take((size_t)HH * EPAD * 2);          // [512,320]
    __bf16* UiouT = (__bf16*)take((size_t)1536 * HH * 2);          // [1536,512]
    __bf16* UfT   = (__bf16*)take((size_t)HH * HH * 2);            // [512,512]
    float*  iou   = (float*)take((size_t)NN * 3 * HH * 4);         // [8192,1536]
    float*  xWf   = (float*)take((size_t)NN * HH * 4);             // [8192,512]
    __bf16* hch   = (__bf16*)take((size_t)NN * KK * HH * 2);       // [32768,512]
    float*  fpre  = (float*)take((size_t)NN * KK * HH * 4);        // [32768,512]
    __bf16* htil  = (__bf16*)take((size_t)NN * HH * 2);            // [8192,512]
    float*  fcbuf = (float*)take((size_t)NN * HH * 4);             // [8192,512]
    float*  hA    = (float*)take((size_t)NN * HH * 4);
    float*  cA    = (float*)take((size_t)NN * HH * 4);
    float*  hB    = (float*)take((size_t)NN * HH * 4);
    float*  cB    = (float*)take((size_t)NN * HH * 4);

    // ---- weights -> transposed bf16 (done each launch; cheap, deterministic)
    transpose_convert_kernel<<<(1536 * EPAD + 255) / 256, 256, 0, stream>>>(
        W_iou, WiouT, EE, 1536, EPAD);
    transpose_convert_kernel<<<(HH * EPAD + 255) / 256, 256, 0, stream>>>(
        W_f, WfT, EE, HH, EPAD);
    transpose_convert_kernel<<<(1536 * HH + 255) / 256, 256, 0, stream>>>(
        U_iou, UiouT, HH, 1536, HH);
    transpose_convert_kernel<<<(HH * HH + 255) / 256, 256, 0, stream>>>(
        U_f, UfT, HH, HH, HH);

    float* hPrev = hA; float* cPrev = cA;
    float* hCur  = hB; float* cCur  = cB;

    for (int l = 5; l >= 0; --l) {
        // x = embed[ix] * mask  (bf16, zero-padded to K=320)
        gather_embed_kernel<<<NN, 128, 0, stream>>>(
            vocab_ix + l * NN, token_mask + l * NN, embed, Xbf);

        // iou = x @ W_iou + b_iou
        {
            dim3 g(NN / 128, (3 * HH) / 128);
            gemm_bf16_wmma_kernel<<<g, 256, 0, stream>>>(
                Xbf, WiouT, iou, b_iou, NN, 3 * HH, EPAD, 0);
        }
        // xWf = x @ W_f   (b_f added later inside fuse_fc)
        {
            dim3 g(NN / 128, HH / 128);
            gemm_bf16_wmma_kernel<<<g, 256, 0, stream>>>(
                Xbf, WfT, xWf, nullptr, NN, HH, EPAD, 0);
        }

        if (l < 5) {
            gather_children_kernel<<<NN, 256, 0, stream>>>(
                child_idx + (size_t)l * NN * KK, child_mask + (size_t)l * NN * KK,
                hPrev, hch, htil);
            // fpre = h_ch @ U_f   (M = N*K = 32768)
            {
                dim3 g((NN * KK) / 128, HH / 128);
                gemm_bf16_wmma_kernel<<<g, 256, 0, stream>>>(
                    hch, UfT, fpre, nullptr, NN * KK, HH, HH, 0);
            }
            fuse_fc_kernel<<<NN, 256, 0, stream>>>(
                child_idx + (size_t)l * NN * KK, child_mask + (size_t)l * NN * KK,
                cPrev, xWf, fpre, b_f, fcbuf);
            // iou += h_tilde @ U_iou
            {
                dim3 g(NN / 128, (3 * HH) / 128);
                gemm_bf16_wmma_kernel<<<g, 256, 0, stream>>>(
                    htil, UiouT, iou, nullptr, NN, 3 * HH, HH, 1);
            }
        } else {
            zero_f32_kernel<<<4096, 256, 0, stream>>>(fcbuf, NN * HH);
        }

        float* hOut = (l == 0) ? (float*)d_out : hCur;
        gates_kernel<<<(NN * HH) / 256, 256, 0, stream>>>(iou, fcbuf, cCur, hOut);

        float* t;
        t = hPrev; hPrev = hCur; hCur = t;
        t = cPrev; cPrev = cCur; cCur = t;
    }
}
